// Llama_11227044512153
// MI455X (gfx1250) — matrix operand order repaired and checked by reading, not compiled, not run on hardware
//
#include <hip/hip_runtime.h>
#include <hip/hip_bf16.h>
#include <math.h>
#include <stdint.h>

#define S_    2048
#define D_    2048
#define QKVN_ 3072
#define HQ_   16
#define HK_   4
#define HD_   128
#define G_    4
#define P_    8192
#define V_    32000
#define L_    2
#define EPS_  1e-5f
#define NEG_  -1e30f
#define THETA_ 500000.0f

typedef __attribute__((ext_vector_type(16))) __bf16 v16bf;
typedef __attribute__((ext_vector_type(8)))  __bf16 v8bf;
typedef __attribute__((ext_vector_type(8)))  float  v8f;
typedef __attribute__((ext_vector_type(4)))  float  v4f;

// ---------------------------------------------------------------------------
// WMMA 16x16x32 bf16 fragment layouts used here (contiguous k order; C per ISA 7.12.2, wave32):
// A: lane<16 -> row=lane,    K elements k0..k0+15
//    lane>=16 -> row=lane-16, K elements k0+16..k0+31
// B: lane<16 -> col=lane,    elements e = K k0+e (0..15)
//    lane>=16 -> col=lane-16, elements e = K k0+16+e
// C: lane&15 = column, VGPR r = row r (lane<16) / r+8 (lane>=16)
// Weights / K / V are pre-swizzled so a B fragment is one 32B load per lane.
// ---------------------------------------------------------------------------

static __device__ inline v16bf load_a_bf16(const __bf16* __restrict__ A, int lda,
                                           int row0, int k0, int lane) {
  int m  = lane & 15;
  int ko = (lane < 16) ? 0 : 16;
  const __bf16* p = A + (size_t)(row0 + m) * lda + k0 + ko;
  v8bf lo = *(const v8bf*)p;         // K ko..ko+7
  v8bf hi = *(const v8bf*)(p + 8);   // K ko+8..ko+15
  return __builtin_shufflevector(lo, hi, 0, 1, 2, 3, 4, 5, 6, 7,
                                 8, 9, 10, 11, 12, 13, 14, 15);
}

static __device__ inline float half_max(float v) {
#pragma unroll
  for (int m = 1; m <= 8; m <<= 1) v = fmaxf(v, __shfl_xor(v, m, 32));
  return v;
}
static __device__ inline float half_sum(float v) {
#pragma unroll
  for (int m = 1; m <= 8; m <<= 1) v += __shfl_xor(v, m, 32);
  return v;
}

// --- CDNA5 async copy global->LDS (ASYNCcnt), CDNA5 ISA ----
// Copies 32 bytes per call; INST_OFFSET applies to both global and LDS sides.
static __device__ inline void async_copy32(uint32_t lds_off, const void* gsrc) {
  asm volatile("global_load_async_to_lds_b128 %0, %1, off"
               :: "v"(lds_off), "v"(gsrc) : "memory");
  asm volatile("global_load_async_to_lds_b128 %0, %1, off offset:16"
               :: "v"(lds_off), "v"(gsrc) : "memory");
}
static __device__ inline void wait_async0() {
  asm volatile("s_wait_asynccnt 0x0" ::: "memory");
}

// ---------------------------------------------------------------------------
// Weight swizzle: B[KxN] fp32 row-major -> bf16 fragment-major.
// Fragment tile (tk, tn) occupies 512 elements at ((tn*Kt + tk)*32 + lane)*16,
// each lane's 16 elements contiguous. One thread per (tile, lane).
// ---------------------------------------------------------------------------
__global__ __launch_bounds__(256) void k_swizzle(const float* __restrict__ B,
                                                 __bf16* __restrict__ O,
                                                 int K, int N) {
  int Kt = K >> 5;
  size_t tid  = (size_t)blockIdx.x * 256 + threadIdx.x;
  int    lane = (int)(tid & 31);
  size_t tile = tid >> 5;
  int tk = (int)(tile % (size_t)Kt);
  int tn = (int)(tile / (size_t)Kt);
  int row = tk * 32 + ((lane & 16) ? 16 : 0);
  int col = tn * 16 + (lane & 15);
  const float* p = B + (size_t)row * N + col;
  v16bf b;
#pragma unroll
  for (int e = 0; e < 16; ++e) b[e] = (__bf16)p[(size_t)e * N];
  *((v16bf*)O + tile * 32 + lane) = b;
}

// V swizzle: per kv-head [S x HD] slice of QKV -> bf16 fragment-major.
// (GEMM K-dim = keys, N-dim = hd columns.)
__global__ __launch_bounds__(256) void k_swizzle_v(const float* __restrict__ QKV,
                                                   __bf16* __restrict__ Vsw) {
  const int Kt = S_ >> 5;   // 64 key tiles
  const int Nt = HD_ >> 4;  // 8 col tiles
  size_t tid  = (size_t)blockIdx.x * 256 + threadIdx.x;
  int    lane = (int)(tid & 31);
  size_t tile = tid >> 5;
  int hk = (int)(tile / (size_t)(Kt * Nt));
  int t2 = (int)(tile % (size_t)(Kt * Nt));
  int tk = t2 % Kt;
  int tn = t2 / Kt;
  int row = tk * 32 + ((lane & 16) ? 16 : 0);
  int col = tn * 16 + (lane & 15);
  const float* p = QKV + (size_t)row * QKVN_ + (D_ + HK_ * HD_) + hk * HD_ + col;
  v16bf b;
#pragma unroll
  for (int e = 0; e < 16; ++e) b[e] = (__bf16)p[(size_t)e * QKVN_];
  *((v16bf*)Vsw + tile * 32 + lane) = b;
}

// K swizzle: per kv-head, K^T as B fragments (GEMM K-dim = hd, N-dim = keys).
// Fragment (tn=key tile, tk=hd tile): lane col = key, elements = hd contiguous.
__global__ __launch_bounds__(256) void k_swizzle_k(const float* __restrict__ QKV,
                                                   __bf16* __restrict__ Ksw) {
  const int Kt = HD_ >> 5;  // 4 hd tiles
  const int Nt = S_ >> 4;   // 128 key tiles
  size_t tid  = (size_t)blockIdx.x * 256 + threadIdx.x;
  int    lane = (int)(tid & 31);
  size_t tile = tid >> 5;
  int hk = (int)(tile / (size_t)(Kt * Nt));
  int t2 = (int)(tile % (size_t)(Kt * Nt));
  int tk = t2 % Kt;                      // hd tile
  int tn = t2 / Kt;                      // key tile
  int key = tn * 16 + (lane & 15);
  int k0  = tk * 32 + ((lane & 16) ? 16 : 0);
  const float* p = QKV + (size_t)key * QKVN_ + D_ + hk * HD_ + k0;
  v4f a0 = *(const v4f*)p;
  v4f a1 = *(const v4f*)(p + 4);
  v4f a2 = *(const v4f*)(p + 8);
  v4f a3 = *(const v4f*)(p + 12);
  v16bf b;
#pragma unroll
  for (int j = 0; j < 4; ++j) {
    b[j]      = (__bf16)a0[j];
    b[4 + j]  = (__bf16)a1[j];
    b[8 + j]  = (__bf16)a2[j];
    b[12 + j] = (__bf16)a3[j];
  }
  *((v16bf*)Ksw + tile * 32 + lane) = b;
}

// Convert the Q section of QKV (after RoPE) to bf16 row-major [S, D].
__global__ __launch_bounds__(256) void k_cvt_q(const float* __restrict__ QKV,
                                               __bf16* __restrict__ Qb) {
  int s = blockIdx.x;
  const float* src = QKV + (size_t)s * QKVN_;
  __bf16* dst = Qb + (size_t)s * D_;
  for (int d = threadIdx.x * 4; d < D_; d += 1024) {
    v4f v = *(const v4f*)(src + d);
#pragma unroll
    for (int j = 0; j < 4; ++j) dst[d + j] = (__bf16)v[j];
  }
}

// ---------------------------------------------------------------------------
__global__ __launch_bounds__(256) void k_embed(const int* __restrict__ tokens,
                                               const float* __restrict__ embed,
                                               float* __restrict__ X) {
  int s = blockIdx.x;
  int t = tokens[s];
  const v4f* src = (const v4f*)(embed + (size_t)t * D_);
  v4f* dst = (v4f*)(X + (size_t)s * D_);
  for (int d = threadIdx.x; d < D_ / 4; d += 256) dst[d] = src[d];
}

// RMSNorm with bf16 output (feeds WMMA A operand).
__global__ __launch_bounds__(256) void k_rmsnorm(const float* __restrict__ X,
                                                 const float* __restrict__ w,
                                                 __bf16* __restrict__ O) {
  __shared__ float red[8];
  __shared__ float invs;
  int s = blockIdx.x;
  const float* x = X + (size_t)s * D_;
  float ss = 0.f;
  for (int d = threadIdx.x; d < D_; d += 256) { float v = x[d]; ss = fmaf(v, v, ss); }
#pragma unroll
  for (int m = 1; m < 32; m <<= 1) ss += __shfl_xor(ss, m, 32);
  if ((threadIdx.x & 31) == 0) red[threadIdx.x >> 5] = ss;
  __syncthreads();
  if (threadIdx.x == 0) {
    float t = 0.f;
#pragma unroll
    for (int i = 0; i < 8; ++i) t += red[i];
    invs = rsqrtf(t / (float)D_ + EPS_);
  }
  __syncthreads();
  float inv = invs;
  __bf16* o = O + (size_t)s * D_;
  for (int d = threadIdx.x; d < D_; d += 256) o[d] = (__bf16)(x[d] * inv * w[d]);
}

__global__ __launch_bounds__(256) void k_rope(float* __restrict__ QKV) {
  int s = blockIdx.x;
  float* row = QKV + (size_t)s * QKVN_;
  const int NP = (HQ_ + HK_) * (HD_ / 2);
  const float l2t = log2f(THETA_);
  for (int i = threadIdx.x; i < NP; i += 256) {
    int head = i >> 6;
    int j    = i & 63;
    float* base = (head < HQ_) ? (row + head * HD_)
                               : (row + D_ + (head - HQ_) * HD_);
    float inv = exp2f(-((float)(2 * j) / (float)HD_) * l2t);
    float ang = (float)s * inv;
    float c = cosf(ang), sn = sinf(ang);
    float x0 = base[2 * j], x1 = base[2 * j + 1];
    base[2 * j]     = x0 * c - x1 * sn;
    base[2 * j + 1] = x0 * sn + x1 * c;
  }
}

// ---------------------------------------------------------------------------
// GEMM: C[MxN] f32 = A[MxK] bf16 @ Bsw (pre-swizzled bf16) (+ R).
// Block = 8 waves, tile 128x128. B fragments for the whole block (8KB/K-step)
// staged into LDS via async-copy, double-buffered. Per K-step: clause of 16
// ds_load_b128 into registers, then 8 back-to-back WMMAs (independent accs).
// ---------------------------------------------------------------------------
__global__ __launch_bounds__(256) void k_gemm(const __bf16* __restrict__ A,
                                              const __bf16* __restrict__ Bsw,
                                              float* __restrict__ C,
                                              const float* __restrict__ R,
                                              int K, int lda, int ldc) {
  __shared__ __attribute__((aligned(32))) char smem[16384];  // 2 x 8KB
  int tid  = threadIdx.x;
  int lane = tid & 31;
  int wave = tid >> 5;
  int row0 = blockIdx.y * 128 + wave * 16;
  int col0 = blockIdx.x * 128;
  int Kt   = K >> 5;
  int tn0  = col0 >> 4;
  const v16bf* Bp = (const v16bf*)Bsw;
  uint32_t lds0 = (uint32_t)(uintptr_t)&smem[0];

  int st = tid >> 5;
  int sl = tid & 31;
  auto stage = [&](int tk, int buf) {
    const void* g = (const void*)&Bp[((size_t)(tn0 + st) * Kt + tk) * 32 + sl];
    async_copy32(lds0 + (uint32_t)buf * 8192u + (uint32_t)tid * 32u, g);
  };

  v8f acc[8] = {};
  stage(0, 0);
  v16bf a_cur = load_a_bf16(A, lda, row0, 0, lane);
  for (int tk = 0; tk < Kt; ++tk) {
    wait_async0();
    __syncthreads();
    if (tk + 1 < Kt) stage(tk + 1, (tk + 1) & 1);
    v16bf a_next = a_cur;
    if (tk + 1 < Kt) a_next = load_a_bf16(A, lda, row0, (tk + 1) * 32, lane);
    const v16bf* lb = (const v16bf*)(smem + (size_t)(tk & 1) * 8192);
    v16bf bfr[8];
#pragma unroll
    for (int t = 0; t < 8; ++t) bfr[t] = lb[t * 32 + lane];
#pragma unroll
    for (int t = 0; t < 8; ++t)
      acc[t] = __builtin_amdgcn_wmma_f32_16x16x32_bf16(false, a_cur, false, bfr[t],
                                                       (short)0, acc[t], false, false);
    a_cur = a_next;
  }
  int n  = lane & 15;
  int ro = (lane < 16) ? 0 : 8;
#pragma unroll
  for (int t = 0; t < 8; ++t)
#pragma unroll
    for (int r = 0; r < 8; ++r) {
      size_t idx = (size_t)(row0 + ro + r) * ldc + (col0 + 16 * t + n);
      float v = acc[t][r];
      if (R) v += R[idx];
      C[idx] = v;
    }
}

// Fused MLP-in GEMM: gate/up halves of pre-swizzled Wi, bf16 silu(g)*u output.
// Same async-LDS pipeline; LDS slots 0..3 = gate frags, 4..7 = up frags.
__global__ __launch_bounds__(256) void k_gemm_mlp(const __bf16* __restrict__ A,
                                                  const __bf16* __restrict__ Bsw,
                                                  __bf16* __restrict__ C,
                                                  int K, int lda) {
  __shared__ __attribute__((aligned(32))) char smem[16384];
  int tid  = threadIdx.x;
  int lane = tid & 31;
  int wave = tid >> 5;
  int row0 = blockIdx.y * 128 + wave * 16;
  int col0 = blockIdx.x * 64;
  int Kt   = K >> 5;
  int tng  = col0 >> 4;
  int tnu  = tng + (P_ >> 4);
  const v16bf* Bp = (const v16bf*)Bsw;
  uint32_t lds0 = (uint32_t)(uintptr_t)&smem[0];

  int st  = tid >> 5;
  int sl  = tid & 31;
  int stn = (st < 4) ? (tng + st) : (tnu + st - 4);
  auto stage = [&](int tk, int buf) {
    const void* g = (const void*)&Bp[((size_t)stn * Kt + tk) * 32 + sl];
    async_copy32(lds0 + (uint32_t)buf * 8192u + (uint32_t)tid * 32u, g);
  };

  v8f accg[4] = {};
  v8f accu[4] = {};
  stage(0, 0);
  v16bf a_cur = load_a_bf16(A, lda, row0, 0, lane);
  for (int tk = 0; tk < Kt; ++tk) {
    wait_async0();
    __syncthreads();
    if (tk + 1 < Kt) stage(tk + 1, (tk + 1) & 1);
    v16bf a_next = a_cur;
    if (tk + 1 < Kt) a_next = load_a_bf16(A, lda, row0, (tk + 1) * 32, lane);
    const v16bf* lb = (const v16bf*)(smem + (size_t)(tk & 1) * 8192);
    v16bf bfr[8];
#pragma unroll
    for (int t = 0; t < 8; ++t) bfr[t] = lb[t * 32 + lane];
#pragma unroll
    for (int t = 0; t < 4; ++t) {
      accg[t] = __builtin_amdgcn_wmma_f32_16x16x32_bf16(false, a_cur, false, bfr[t],
                                                        (short)0, accg[t], false, false);
      accu[t] = __builtin_amdgcn_wmma_f32_16x16x32_bf16(false, a_cur, false, bfr[4 + t],
                                                        (short)0, accu[t], false, false);
    }
    a_cur = a_next;
  }
  int n  = lane & 15;
  int ro = (lane < 16) ? 0 : 8;
#pragma unroll
  for (int t = 0; t < 4; ++t)
#pragma unroll
    for (int r = 0; r < 8; ++r) {
      float g = accg[t][r];
      float u = accu[t][r];
      float silu = g / (1.f + expf(-g));
      C[(size_t)(row0 + ro + r) * P_ + (col0 + 16 * t + n)] = (__bf16)(silu * u);
    }
}

// ---------------------------------------------------------------------------
// Flash attention: one wave per (16-query block, head). Q bf16 A-frags hoisted;
// K^T and V fragments pre-swizzled bf16 -> single 32B loads, zero converts.
// ---------------------------------------------------------------------------
__global__ __launch_bounds__(32) void k_attn(const __bf16* __restrict__ Qb,
                                             const __bf16* __restrict__ Ksw,
                                             const __bf16* __restrict__ Vsw,
                                             __bf16* __restrict__ ATT) {
  __shared__ __attribute__((aligned(16))) __bf16 pl[16 * 32];
  int lane = threadIdx.x;
  int qb = blockIdx.x * 16;
  int h  = blockIdx.y;
  int hk = h / G_;
  const int KKt = HD_ >> 5;  // 4 hd tiles (score GEMM K-dim)
  const int VKt = S_ >> 5;   // 64 key tiles (PV GEMM K-dim)
  const v16bf* Kp = (const v16bf*)Ksw + (size_t)hk * (KKt * (S_ >> 4)) * 32;
  const v16bf* Vp = (const v16bf*)Vsw + (size_t)hk * (VKt * (HD_ >> 4)) * 32;
  const float scale = 0.08838834764831845f;  // 1/sqrt(128)

  v16bf qa[4];
#pragma unroll
  for (int kk = 0; kk < 4; ++kk)
    qa[kk] = load_a_bf16(Qb + h * HD_, D_, qb, kk * 32, lane);

  v8f o[8] = {};
  float rmax[8], rsum[8];
#pragma unroll
  for (int r = 0; r < 8; ++r) { rmax[r] = NEG_; rsum[r] = 0.f; }
  int ro = (lane < 16) ? 0 : 8;
  int n  = lane & 15;

  int nch = (qb + 16 + 31) >> 5;
  for (int ci = 0; ci < nch; ++ci) {
    int c0 = ci * 32;
    int tn = c0 >> 4;
    v8f s0 = {}, s1 = {};
    v16bf kb[8];
#pragma unroll
    for (int kk = 0; kk < 4; ++kk) {
      kb[kk]     = Kp[((size_t)tn * KKt + kk) * 32 + lane];
      kb[4 + kk] = Kp[((size_t)(tn + 1) * KKt + kk) * 32 + lane];
    }
#pragma unroll
    for (int kk = 0; kk < 4; ++kk) {
      s0 = __builtin_amdgcn_wmma_f32_16x16x32_bf16(false, qa[kk], false, kb[kk],     (short)0, s0, false, false);
      s1 = __builtin_amdgcn_wmma_f32_16x16x32_bf16(false, qa[kk], false, kb[4 + kk], (short)0, s1, false, false);
    }
    float nmax[8];
#pragma unroll
    for (int r = 0; r < 8; ++r) {
      int row = qb + ro + r;
      float v0 = (c0 + n      <= row) ? s0[r] * scale : NEG_;
      float v1 = (c0 + 16 + n <= row) ? s1[r] * scale : NEG_;
      s0[r] = v0; s1[r] = v1;
      float m = half_max(fmaxf(v0, v1));
      nmax[r] = fmaxf(rmax[r], m);
    }
#pragma unroll
    for (int r = 0; r < 8; ++r) {
      float sc = expf(rmax[r] - nmax[r]);
      float p0 = expf(s0[r] - nmax[r]);
      float p1 = expf(s1[r] - nmax[r]);
      s0[r] = p0; s1[r] = p1;
      rsum[r] = rsum[r] * sc + half_sum(p0 + p1);
      rmax[r] = nmax[r];
#pragma unroll
      for (int t = 0; t < 8; ++t) o[t][r] *= sc;
    }
    // transpose P (C-layout) -> A-layout via LDS
    __syncthreads();
#pragma unroll
    for (int r = 0; r < 8; ++r) {
      pl[(ro + r) * 32 + n]      = (__bf16)s0[r];
      pl[(ro + r) * 32 + 16 + n] = (__bf16)s1[r];
    }
    __syncthreads();
    v16bf pa;
    {
      int m  = lane & 15;
      int co = (lane < 16) ? 0 : 16;
      v8bf lo = *(const v8bf*)(pl + m * 32 + co);
      v8bf hi = *(const v8bf*)(pl + m * 32 + 8 + co);
      pa = __builtin_shufflevector(lo, hi, 0, 1, 2, 3, 4, 5, 6, 7,
                                   8, 9, 10, 11, 12, 13, 14, 15);
    }
    // O += P @ V from pre-swizzled V fragments
    int tk = c0 >> 5;
    v16bf vb[8];
#pragma unroll
    for (int t = 0; t < 8; ++t) vb[t] = Vp[((size_t)t * VKt + tk) * 32 + lane];
#pragma unroll
    for (int t = 0; t < 8; ++t)
      o[t] = __builtin_amdgcn_wmma_f32_16x16x32_bf16(false, pa, false, vb[t], (short)0, o[t], false, false);
  }
#pragma unroll
  for (int t = 0; t < 8; ++t)
#pragma unroll
    for (int r = 0; r < 8; ++r) {
      int row = qb + ro + r;
      ATT[(size_t)row * D_ + h * HD_ + 16 * t + n] = (__bf16)(o[t][r] / rsum[r]);
    }
}

// ---------------------------------------------------------------------------
extern "C" void kernel_launch(void* const* d_in, const int* in_sizes, int n_in,
                              void* d_out, int out_size, void* d_ws, size_t ws_size,
                              hipStream_t stream) {
  (void)in_sizes; (void)n_in; (void)out_size; (void)ws_size;
  const int*   tokens       = (const int*)d_in[0];
  const float* embed        = (const float*)d_in[1];
  const float* attn_norm_w  = (const float*)d_in[2];
  const float* qkv_w        = (const float*)d_in[3];
  const float* attn_out_w   = (const float*)d_in[4];
  const float* mlp_norm_w   = (const float*)d_in[5];
  const float* mlp_in_w     = (const float*)d_in[6];
  const float* mlp_out_w    = (const float*)d_in[7];
  const float* final_norm_w = (const float*)d_in[8];
  const float* lm_head_w    = (const float*)d_in[9];
  float* out = (float*)d_out;

  // Workspace layout (256B-aligned chunks)
  char* p = (char*)d_ws;
  auto take = [&](size_t bytes) { char* q = p; p += (bytes + 255) & ~(size_t)255; return q; };
  __bf16* W    = (__bf16*)take((size_t)D_ * V_ * 2);       // swizzled weight (max size, reused)
  float*  X    = (float*) take((size_t)S_ * D_ * 4);       // residual
  float*  QKV  = (float*) take((size_t)S_ * QKVN_ * 4);
  __bf16* Hn   = (__bf16*)take((size_t)S_ * D_ * 2);       // normed activations
  __bf16* ATT  = (__bf16*)take((size_t)S_ * D_ * 2);       // attention output
  __bf16* MLPA = (__bf16*)take((size_t)S_ * P_ * 2);       // silu(gate)*up
  __bf16* Qb   = (__bf16*)take((size_t)S_ * D_ * 2);       // bf16 Q after rope
  __bf16* Ksw  = (__bf16*)take((size_t)S_ * HK_ * HD_ * 2);
  __bf16* Vsw  = (__bf16*)take((size_t)S_ * HK_ * HD_ * 2);

  dim3 blk256(256), blk32(32);
  auto swz_blocks = [](size_t K, size_t N) { return (unsigned)(K * N / 16 / 256); };
  const unsigned kv_swz_blocks = (HK_ * (S_ / 32) * (HD_ / 16) * 32) / 256;

  k_embed<<<dim3(S_), blk256, 0, stream>>>(tokens, embed, X);

  for (int i = 0; i < L_; ++i) {
    const float* wqkv = qkv_w      + (size_t)i * D_ * QKVN_;
    const float* wo   = attn_out_w + (size_t)i * D_ * D_;
    const float* wi   = mlp_in_w   + (size_t)i * D_ * 2 * P_;
    const float* wmo  = mlp_out_w  + (size_t)i * P_ * D_;

    k_rmsnorm<<<dim3(S_), blk256, 0, stream>>>(X, attn_norm_w + (size_t)i * D_, Hn);
    k_swizzle<<<dim3(swz_blocks(D_, QKVN_)), blk256, 0, stream>>>(wqkv, W, D_, QKVN_);
    k_gemm<<<dim3(QKVN_ / 128, S_ / 128), blk256, 0, stream>>>(
        Hn, W, QKV, nullptr, D_, D_, QKVN_);
    k_rope<<<dim3(S_), blk256, 0, stream>>>(QKV);
    k_cvt_q<<<dim3(S_), blk256, 0, stream>>>(QKV, Qb);
    k_swizzle_k<<<dim3(kv_swz_blocks), blk256, 0, stream>>>(QKV, Ksw);
    k_swizzle_v<<<dim3(kv_swz_blocks), blk256, 0, stream>>>(QKV, Vsw);
    k_attn<<<dim3(S_ / 16, HQ_), blk32, 0, stream>>>(Qb, Ksw, Vsw, ATT);
    k_swizzle<<<dim3(swz_blocks(D_, D_)), blk256, 0, stream>>>(wo, W, D_, D_);
    k_gemm<<<dim3(D_ / 128, S_ / 128), blk256, 0, stream>>>(
        ATT, W, X, X, D_, D_, D_);  // X += ATT @ Wo

    k_rmsnorm<<<dim3(S_), blk256, 0, stream>>>(X, mlp_norm_w + (size_t)i * D_, Hn);
    k_swizzle<<<dim3(swz_blocks(D_, 2 * P_)), blk256, 0, stream>>>(wi, W, D_, 2 * P_);
    k_gemm_mlp<<<dim3(P_ / 64, S_ / 128), blk256, 0, stream>>>(Hn, W, MLPA, D_, D_);
    k_swizzle<<<dim3(swz_blocks(P_, D_)), blk256, 0, stream>>>(wmo, W, P_, D_);
    k_gemm<<<dim3(D_ / 128, S_ / 128), blk256, 0, stream>>>(
        MLPA, W, X, X, P_, P_, D_);  // X += MLPA @ Wmo
  }

  k_rmsnorm<<<dim3(S_), blk256, 0, stream>>>(X, final_norm_w, Hn);
  k_swizzle<<<dim3(swz_blocks(D_, V_)), blk256, 0, stream>>>(lm_head_w, W, D_, V_);
  k_gemm<<<dim3(V_ / 128, S_ / 128), blk256, 0, stream>>>(
      Hn, W, out, nullptr, D_, D_, V_);
}
